// Model_3582002725443
// MI455X (gfx1250) — compile-verified, matrix operand order checked by reading, not run on hardware
//
#include <hip/hip_runtime.h>
#include <hip/hip_bf16.h>
#include <math.h>

// ---------------------------------------------------------------------------
// Problem constants (from reference)
// ---------------------------------------------------------------------------
#define T_SEQ 96
#define B_DIM 64
#define DM    1024
#define DG    512
#define HD    256                  // LSTM hidden per direction
#define M_ROWS (T_SEQ * B_DIM)     // 6144 GEMM rows
#define KC    128                  // K-chunk staged to LDS by the TDM

typedef __attribute__((ext_vector_type(16))) __bf16 v16bf;
typedef __attribute__((ext_vector_type(8)))  float  v8f;
typedef __attribute__((ext_vector_type(4)))  unsigned int v4u;
typedef __attribute__((ext_vector_type(8)))  int    v8i;
typedef __attribute__((ext_vector_type(4)))  int    v4i;

// ---------------------------------------------------------------------------
// Helpers
// ---------------------------------------------------------------------------
__device__ __forceinline__ unsigned short f2bf(float x) {
  unsigned int u = __builtin_bit_cast(unsigned int, x);
  u += 0x7FFFu + ((u >> 16) & 1u);           // round-to-nearest-even
  return (unsigned short)(u >> 16);
}

// Load one 16x32 bf16 WMMA fragment (A or B operand) for this lane.
// Lane L holds row (L%16); elements 0..7 = K[8*(L/16)..+8),
// elements 8..15 = K[16+8*(L/16)..+8).  p = base + row*ld + k0 + 8*(lane>>4).
__device__ __forceinline__ v16bf load_frag_bf(const unsigned short* p) {
  v16bf f;
#pragma unroll
  for (int j = 0; j < 8; ++j) f[j]     = __builtin_bit_cast(__bf16, p[j]);
#pragma unroll
  for (int j = 0; j < 8; ++j) f[j + 8] = __builtin_bit_cast(__bf16, p[16 + j]);
  return f;
}

__device__ __forceinline__ float sigf(float x) { return 1.0f / (1.0f + __expf(-x)); }

// ---------------------------------------------------------------------------
// 1) LN2 statistics: per (input i, batch b) mean & rstd over T*DM elements
// ---------------------------------------------------------------------------
__global__ void ln2_stats_kernel(const float* __restrict__ r1, const float* __restrict__ r2,
                                 const float* __restrict__ r3, const float* __restrict__ r4,
                                 float* __restrict__ stats) {
  const float* rs[4] = {r1, r2, r3, r4};
  const float* r = rs[blockIdx.y];
  int b = blockIdx.x;
  float s = 0.f, sq = 0.f;
  for (int t = 0; t < T_SEQ; ++t) {
    const float* row = r + (size_t)(t * B_DIM + b) * DM;
    for (int d = threadIdx.x; d < DM; d += blockDim.x) {
      float v = row[d];
      s += v; sq += v * v;
    }
  }
  __shared__ float ss[256], sqq[256];
  ss[threadIdx.x] = s; sqq[threadIdx.x] = sq;
  __syncthreads();
  for (int off = 128; off > 0; off >>= 1) {
    if ((int)threadIdx.x < off) {
      ss[threadIdx.x]  += ss[threadIdx.x + off];
      sqq[threadIdx.x] += sqq[threadIdx.x + off];
    }
    __syncthreads();
  }
  if (threadIdx.x == 0) {
    float n = (float)(T_SEQ * DM);
    float m = ss[0] / n;
    float v = sqq[0] / n - m * m;
    stats[(blockIdx.y * B_DIM + b) * 2 + 0] = m;
    stats[(blockIdx.y * B_DIM + b) * 2 + 1] = rsqrtf(v + 1e-5f);
  }
}

// ---------------------------------------------------------------------------
// 2) U = mean of the 4 normalized tensors, emitted as bf16 [M_ROWS, DM]
// ---------------------------------------------------------------------------
__global__ void ln2_combine_kernel(const float* __restrict__ r1, const float* __restrict__ r2,
                                   const float* __restrict__ r3, const float* __restrict__ r4,
                                   const float* __restrict__ stats,
                                   unsigned short* __restrict__ Ubf) {
  unsigned idx = blockIdx.x * blockDim.x + threadIdx.x;
  if (idx >= M_ROWS * DM) return;
  unsigned b = (idx >> 10) & 63;             // (idx / DM) % B_DIM, DM=1024
  const float* rs[4] = {r1, r2, r3, r4};
  float acc = 0.f;
#pragma unroll
  for (int i = 0; i < 4; ++i) {
    float m  = stats[(i * B_DIM + b) * 2 + 0];
    float is = stats[(i * B_DIM + b) * 2 + 1];
    acc += (rs[i][idx] - m) * is;
  }
  Ubf[idx] = f2bf(acc * 0.25f);
}

// ---------------------------------------------------------------------------
// 3) f32 [R,K] -> bf16 [R,Kp] (zero-padded in K)
// ---------------------------------------------------------------------------
__global__ void pack_bf16_kernel(const float* __restrict__ src, unsigned short* __restrict__ dst,
                                 int R, int K, int Kp) {
  int idx = blockIdx.x * blockDim.x + threadIdx.x;
  if (idx >= R * Kp) return;
  int r = idx / Kp, k = idx % Kp;
  float v = (k < K) ? src[(size_t)r * K + k] : 0.f;
  dst[idx] = f2bf(v);
}

__global__ void add2_kernel(const float* __restrict__ a, const float* __restrict__ b,
                            float* __restrict__ o, int n) {
  int i = blockIdx.x * blockDim.x + threadIdx.x;
  if (i < n) o[i] = a[i] + b[i];
}

// ---------------------------------------------------------------------------
// 4) Tiled WMMA GEMM:  C[M,N] = A[M,K](bf16) * W[N,K]^T(bf16) + bias
//    block = 128 threads (4 waves), each wave computes a 32x32 tile (2x2 frags)
//    grid  = (N/64, M/64).
//    Weight tile (64 rows x KC cols) staged into LDS by the Tensor Data Mover
//    (tensor_load_to_lds), double-buffered, pipelined one chunk ahead with
//    s_wait_tensorcnt.  A operand streams from global (L2).
//    OMODE: 0 = f32 graphified (row = (m%64)*96 + m/64), 1 = f32 plain,
//           2 = bf16 plain.
// ---------------------------------------------------------------------------
template <int OMODE>
__global__ void gemm_wmma_kernel(const unsigned short* __restrict__ A,
                                 const unsigned short* __restrict__ W,
                                 const float* __restrict__ bias,
                                 void* __restrict__ outp,
                                 int K, int ldo, int out_off) {
  __shared__ unsigned short ldsW[2][64 * KC];   // 2 x 16 KB

  const int lane = threadIdx.x & 31;
  const int wave = threadIdx.x >> 5;
  const int half = lane >> 4;
  const int nl   = lane & 15;
  const int Mb       = blockIdx.y * 64 + (wave >> 1) * 32;
  const int nb_local = (wave & 1) * 32;
  const int Nb       = blockIdx.x * 64 + nb_local;
  const int nChunks  = K / KC;                  // K in {128,512,1024}

  // --- TDM descriptor issue: one 2-D tile (64 rows x KC bf16) of W ---------
  auto issue_tdm = [&](int c) {
    unsigned lds_addr = (unsigned)(uintptr_t)(&ldsW[c & 1][0]);
    unsigned long long ga = (unsigned long long)(uintptr_t)W +
        ((unsigned long long)((unsigned)blockIdx.x * 64u) * (unsigned)K +
         (unsigned)(c * KC)) * 2ull;
    // D# group0: count=1 | lds_addr | global_addr[56:0] | type=2
    v4u g0 = { 1u,
               lds_addr,
               (unsigned)(ga & 0xFFFFFFFFu),
               ((unsigned)((ga >> 32) & 0x01FFFFFFu)) | 0x80000000u };
    // D# group1: data_size=2B; tensor_dim0=K; tensor_dim1=64 rows;
    //            tile_dim0=KC; tile_dim1=64; tensor_dim0_stride=K
    unsigned d0 = (unsigned)K, d1 = 64u, s0 = (unsigned)K;
    v8i g1 = { (int)(1u << 16),
               (int)((d0 & 0xFFFFu) << 16),
               (int)((d0 >> 16) | ((d1 & 0xFFFFu) << 16)),
               (int)((d1 >> 16) | ((unsigned)KC << 16)),
               (int)64,
               (int)s0,
               0, 0 };
    v4i gz4 = {0, 0, 0, 0};
    v8i gz8 = {0, 0, 0, 0, 0, 0, 0, 0};
    __builtin_amdgcn_tensor_load_to_lds(g0, g1, gz4, gz4, gz8, 0);
  };

  if (wave == 0) issue_tdm(0);

  v8f acc[2][2] = {};
  for (int c = 0; c < nChunks; ++c) {
    if (wave == 0) {
      if (c + 1 < nChunks) {
        issue_tdm(c + 1);                       // prefetch next chunk
        __builtin_amdgcn_s_wait_tensorcnt(1);   // chunk c complete (in-order)
      } else {
        __builtin_amdgcn_s_wait_tensorcnt(0);
      }
    }
    __syncthreads();                            // LDS chunk visible to all waves
    const unsigned short* Wl = &ldsW[c & 1][0];
#pragma unroll 2
    for (int kc = 0; kc < KC; kc += 32) {
      const int k0 = c * KC + kc;
      v16bf a[2], b[2];
#pragma unroll
      for (int tm = 0; tm < 2; ++tm)
        a[tm] = load_frag_bf(A + (size_t)(Mb + tm * 16 + nl) * K + k0 + 8 * half);
#pragma unroll
      for (int tn = 0; tn < 2; ++tn)
        b[tn] = load_frag_bf(Wl + (nb_local + tn * 16 + nl) * KC + kc + 8 * half);
#pragma unroll
      for (int tm = 0; tm < 2; ++tm)
#pragma unroll
        for (int tn = 0; tn < 2; ++tn)
          acc[tm][tn] = __builtin_amdgcn_wmma_f32_16x16x32_bf16(
              false, a[tm], false, b[tn], (short)0, acc[tm][tn], false, false);
    }
    __syncthreads();                            // done reading before rewrite
  }

  // --- epilogue (branch-free, bias hoisted) --------------------------------
  float bv[2];
#pragma unroll
  for (int tn = 0; tn < 2; ++tn) bv[tn] = bias[Nb + tn * 16 + nl];

#pragma unroll
  for (int tm = 0; tm < 2; ++tm)
#pragma unroll
    for (int tn = 0; tn < 2; ++tn) {
#pragma unroll
      for (int j = 0; j < 8; ++j) {
        unsigned m = (unsigned)(Mb + tm * 16 + j + 8 * half);
        unsigned n = (unsigned)(Nb + tn * 16 + nl);
        float v = acc[tm][tn][j] + bv[tn];
        if (OMODE == 0) {
          unsigned row = (m & 63u) * T_SEQ + (m >> 6);   // graphify: b*T + t
          ((float*)outp)[(size_t)row * ldo + out_off + n] = v;
        } else if (OMODE == 1) {
          ((float*)outp)[(size_t)m * ldo + n] = v;
        } else {
          ((unsigned short*)outp)[(size_t)m * ldo + n] = f2bf(v);
        }
      }
    }
}

// ---------------------------------------------------------------------------
// 5) LSTM recurrence (one direction per workgroup; grid = 2 directions).
//    1024 threads = 32 waves. 64 output tiles/step = 4 M-tiles x 16 H-tiles;
//    each wave owns 2 (mt,ht) pairs and all 4 gates for them, so the cell
//    update is register/LDS-local. h double-buffered bf16 in LDS; c f32 in
//    LDS (128 KB dynamic of 320 KB/WGP). Next timestep's xg slab prefetched.
//    MODE: 0 = bf16 out [T,B,DG] (layer 0), 1 = f32 graphified out (layer 1).
// ---------------------------------------------------------------------------
template <int MODE>
__global__ void lstm_recur_kernel(const float* __restrict__ xgF, const float* __restrict__ xgB,
                                  const unsigned short* __restrict__ WhhF,
                                  const unsigned short* __restrict__ WhhB,
                                  unsigned short* __restrict__ outBF,
                                  float* __restrict__ outF, int outF_off) {
  extern __shared__ char smem[];
  unsigned short* hbuf = (unsigned short*)smem;                            // [2][64][256]
  float* cbuf = (float*)(smem + 2 * B_DIM * HD * sizeof(unsigned short));  // [64][256]

  const int dir = blockIdx.x;
  const float* xg = dir ? xgB : xgF;
  const unsigned short* Whh = dir ? WhhB : WhhF;

  const int tid = threadIdx.x;
  for (int i = tid; i < 2 * B_DIM * HD; i += blockDim.x) hbuf[i] = 0;
  for (int i = tid; i < B_DIM * HD;     i += blockDim.x) cbuf[i] = 0.f;
  __syncthreads();

  const int lane = tid & 31, wave = tid >> 5;
  const int half = lane >> 4, nl = lane & 15;

  for (int t = 0; t < T_SEQ; ++t) {
    const int tt  = dir ? (T_SEQ - 1 - t) : t;
    const int cur = t & 1, nxt = cur ^ 1;
    unsigned short* hcur = hbuf + cur * B_DIM * HD;
    unsigned short* hnxt = hbuf + nxt * B_DIM * HD;

    // Prefetch next timestep's xg slab (256 KB) while this step computes.
    if (t + 1 < T_SEQ) {
      const int tp = dir ? (T_SEQ - 2 - t) : (t + 1);
      const float* pf = xg + (size_t)tp * B_DIM * (4 * HD) + (size_t)tid * 64;
      __builtin_prefetch(pf, 0, 1);
      __builtin_prefetch(pf + 32, 0, 1);
    }

#pragma unroll
    for (int q = 0; q < 2; ++q) {
      const int p  = wave * 2 + q;
      const int mt = p & 3;          // batch tile (4 x 16 rows)
      const int ht = p >> 2;         // hidden tile (16 x 16 cols)
      v8f acc[4];
      // C init = xg[t] (x@Wih^T + bih + bhh)
#pragma unroll
      for (int g = 0; g < 4; ++g)
#pragma unroll
        for (int j = 0; j < 8; ++j) {
          int m = mt * 16 + j + 8 * half;
          acc[g][j] = xg[((size_t)tt * B_DIM + m) * (4 * HD) + g * HD + ht * 16 + nl];
        }
      // g += h @ Whh^T  (K = 256, 8 wmma steps; A from LDS, B from L2)
#pragma unroll
      for (int ks = 0; ks < 8; ++ks) {
        const int k0 = ks * 32;
        v16bf a = load_frag_bf(hcur + (mt * 16 + nl) * HD + k0 + 8 * half);
#pragma unroll
        for (int g = 0; g < 4; ++g) {
          v16bf b = load_frag_bf(Whh + (size_t)(g * HD + ht * 16 + nl) * HD + k0 + 8 * half);
          acc[g] = __builtin_amdgcn_wmma_f32_16x16x32_bf16(
              false, a, false, b, (short)0, acc[g], false, false);
        }
      }
      // gate nonlinearities + cell/hidden update (torch order i,f,g,o)
#pragma unroll
      for (int j = 0; j < 8; ++j) {
        const int mrow = mt * 16 + j + 8 * half;
        const int hcol = ht * 16 + nl;
        float iv = sigf(acc[0][j]);
        float fv = sigf(acc[1][j]);
        float gv = tanhf(acc[2][j]);
        float ov = sigf(acc[3][j]);
        float c  = fv * cbuf[mrow * HD + hcol] + iv * gv;
        cbuf[mrow * HD + hcol] = c;
        float h = ov * tanhf(c);
        hnxt[mrow * HD + hcol] = f2bf(h);
        if (MODE == 0) {
          outBF[((size_t)tt * B_DIM + mrow) * DG + dir * HD + hcol] = f2bf(h);
        } else {
          outF[((size_t)mrow * T_SEQ + tt) * 1536 + outF_off + dir * HD + hcol] = h;
        }
      }
    }
    __syncthreads();   // h[nxt] complete before next step reads it
  }
}

// ---------------------------------------------------------------------------
// Host-side orchestration
// ---------------------------------------------------------------------------
extern "C" void kernel_launch(void* const* d_in, const int* in_sizes, int n_in,
                              void* d_out, int out_size, void* d_ws, size_t ws_size,
                              hipStream_t stream) {
  const float* r1  = (const float*)d_in[0];
  const float* r2  = (const float*)d_in[1];
  const float* r3  = (const float*)d_in[2];
  const float* r4  = (const float*)d_in[3];
  const float* U_a = (const float*)d_in[4];
  const float* U_v = (const float*)d_in[5];
  // d_in[6] = seq_lengths (all full; unused)
  const float* W_a = (const float*)d_in[7];
  const float* b_a = (const float*)d_in[8];
  const float* W_v = (const float*)d_in[9];
  const float* b_v = (const float*)d_in[10];
  const float* W_l = (const float*)d_in[11];
  const float* b_l = (const float*)d_in[12];
  const float* Wih0f = (const float*)d_in[13];
  const float* Whh0f = (const float*)d_in[14];
  const float* bih0f = (const float*)d_in[15];
  const float* bhh0f = (const float*)d_in[16];
  const float* Wih0b = (const float*)d_in[17];
  const float* Whh0b = (const float*)d_in[18];
  const float* bih0b = (const float*)d_in[19];
  const float* bhh0b = (const float*)d_in[20];
  const float* Wih1f = (const float*)d_in[21];
  const float* Whh1f = (const float*)d_in[22];
  const float* bih1f = (const float*)d_in[23];
  const float* bhh1f = (const float*)d_in[24];
  const float* Wih1b = (const float*)d_in[25];
  const float* Whh1b = (const float*)d_in[26];
  const float* bih1b = (const float*)d_in[27];
  const float* bhh1b = (const float*)d_in[28];

  float* out = (float*)d_out;   // [6144, 1536]

  // ---- workspace carve-up -------------------------------------------------
  char* ws = (char*)d_ws;
  size_t off = 0;
  auto alloc = [&](size_t bytes) -> char* {
    char* p = ws + off;
    off = (off + bytes + 255) & ~(size_t)255;
    return p;
  };
  float*          stats  = (float*)alloc(4 * B_DIM * 2 * sizeof(float));
  unsigned short* Ubf    = (unsigned short*)alloc((size_t)M_ROWS * DM * 2);
  unsigned short* Uabf   = (unsigned short*)alloc((size_t)M_ROWS * 128 * 2);
  unsigned short* Uvbf   = (unsigned short*)alloc((size_t)M_ROWS * 512 * 2);
  unsigned short* Wabf   = (unsigned short*)alloc((size_t)512 * 128 * 2);
  unsigned short* Wvbf   = (unsigned short*)alloc((size_t)512 * 512 * 2);
  unsigned short* Wlbf   = (unsigned short*)alloc((size_t)512 * 1024 * 2);
  unsigned short* Wih0fb = (unsigned short*)alloc((size_t)1024 * 512 * 2);
  unsigned short* Wih0bb = (unsigned short*)alloc((size_t)1024 * 512 * 2);
  unsigned short* Wih1fb = (unsigned short*)alloc((size_t)1024 * 512 * 2);
  unsigned short* Wih1bb = (unsigned short*)alloc((size_t)1024 * 512 * 2);
  unsigned short* Whh0fb = (unsigned short*)alloc((size_t)1024 * 256 * 2);
  unsigned short* Whh0bb = (unsigned short*)alloc((size_t)1024 * 256 * 2);
  unsigned short* Whh1fb = (unsigned short*)alloc((size_t)1024 * 256 * 2);
  unsigned short* Whh1bb = (unsigned short*)alloc((size_t)1024 * 256 * 2);
  float*          cb0f   = (float*)alloc(1024 * sizeof(float));
  float*          cb0b   = (float*)alloc(1024 * sizeof(float));
  float*          cb1f   = (float*)alloc(1024 * sizeof(float));
  float*          cb1b   = (float*)alloc(1024 * sizeof(float));
  unsigned short* Ulbf   = (unsigned short*)alloc((size_t)M_ROWS * 512 * 2);
  unsigned short* h0bf   = (unsigned short*)alloc((size_t)M_ROWS * 512 * 2);
  float*          xgF    = (float*)alloc((size_t)M_ROWS * 1024 * sizeof(float));
  float*          xgB    = (float*)alloc((size_t)M_ROWS * 1024 * sizeof(float));
  (void)ws_size; (void)in_sizes; (void)n_in; (void)out_size;

  // ---- stage 1: LN2 + mean -> bf16 U -------------------------------------
  ln2_stats_kernel<<<dim3(B_DIM, 4), 256, 0, stream>>>(r1, r2, r3, r4, stats);
  {
    int n = M_ROWS * DM;
    ln2_combine_kernel<<<(n + 255) / 256, 256, 0, stream>>>(r1, r2, r3, r4, stats, Ubf);
  }

  // ---- stage 2: pack weights & activations to bf16 ------------------------
  auto pack = [&](const float* src, unsigned short* dst, int R, int K, int Kp) {
    int n = R * Kp;
    pack_bf16_kernel<<<(n + 255) / 256, 256, 0, stream>>>(src, dst, R, K, Kp);
  };
  pack(U_a,   Uabf,   M_ROWS, 100, 128);
  pack(U_v,   Uvbf,   M_ROWS, 512, 512);
  pack(W_a,   Wabf,   512, 100, 128);
  pack(W_v,   Wvbf,   512, 512, 512);
  pack(W_l,   Wlbf,   512, 1024, 1024);
  pack(Wih0f, Wih0fb, 1024, 512, 512);
  pack(Wih0b, Wih0bb, 1024, 512, 512);
  pack(Wih1f, Wih1fb, 1024, 512, 512);
  pack(Wih1b, Wih1bb, 1024, 512, 512);
  pack(Whh0f, Whh0fb, 1024, 256, 256);
  pack(Whh0b, Whh0bb, 1024, 256, 256);
  pack(Whh1f, Whh1fb, 1024, 256, 256);
  pack(Whh1b, Whh1bb, 1024, 256, 256);
  add2_kernel<<<4, 256, 0, stream>>>(bih0f, bhh0f, cb0f, 1024);
  add2_kernel<<<4, 256, 0, stream>>>(bih0b, bhh0b, cb0b, 1024);
  add2_kernel<<<4, 256, 0, stream>>>(bih1f, bhh1f, cb1f, 1024);
  add2_kernel<<<4, 256, 0, stream>>>(bih1b, bhh1b, cb1b, 1024);

  // ---- stage 3: acoustic / visual linears straight into graphified d_out --
  gemm_wmma_kernel<0><<<dim3(512 / 64, M_ROWS / 64), 128, 0, stream>>>(
      Uabf, Wabf, b_a, out, 128, 1536, 0);                 // cols [0,512)
  gemm_wmma_kernel<0><<<dim3(512 / 64, M_ROWS / 64), 128, 0, stream>>>(
      Uvbf, Wvbf, b_v, out, 512, 1536, 512);               // cols [512,1024)
  gemm_wmma_kernel<2><<<dim3(512 / 64, M_ROWS / 64), 128, 0, stream>>>(
      Ubf, Wlbf, b_l, Ulbf, 1024, 512, 0);                 // Ul (bf16)

  // ---- stage 4: layer-0 biLSTM -------------------------------------------
  gemm_wmma_kernel<1><<<dim3(1024 / 64, M_ROWS / 64), 128, 0, stream>>>(
      Ulbf, Wih0fb, cb0f, xgF, 512, 1024, 0);
  gemm_wmma_kernel<1><<<dim3(1024 / 64, M_ROWS / 64), 128, 0, stream>>>(
      Ulbf, Wih0bb, cb0b, xgB, 512, 1024, 0);
  lstm_recur_kernel<0><<<2, 1024, 131072, stream>>>(
      xgF, xgB, Whh0fb, Whh0bb, h0bf, nullptr, 0);

  // ---- stage 5: layer-1 biLSTM (reuses xg buffers; writes d_out 1024+) ----
  gemm_wmma_kernel<1><<<dim3(1024 / 64, M_ROWS / 64), 128, 0, stream>>>(
      h0bf, Wih1fb, cb1f, xgF, 512, 1024, 0);
  gemm_wmma_kernel<1><<<dim3(1024 / 64, M_ROWS / 64), 128, 0, stream>>>(
      h0bf, Wih1bb, cb1b, xgB, 512, 1024, 0);
  lstm_recur_kernel<1><<<2, 1024, 131072, stream>>>(
      xgF, xgB, Whh1fb, Whh1bb, nullptr, out, 1024);
}
